// DualPath_12403865550987
// MI455X (gfx1250) — matrix operand order repaired and checked by reading, not compiled, not run on hardware
//
#include <hip/hip_runtime.h>

// ---------------------------------------------------------------------------
// DualPath block for MI455X (gfx1250): bf16 WMMA GEMMs + f32 accumulation.
// ---------------------------------------------------------------------------

typedef __attribute__((ext_vector_type(16))) __bf16 bf16x16;
typedef __attribute__((ext_vector_type(8)))  float  f32x8;

union ABFrag { unsigned int u[8]; uint4 q[2]; bf16x16 v; };

#define WMMA_BF16(C, A, B) \
  __builtin_amdgcn_wmma_f32_16x16x32_bf16(false, (A), false, (B), (short)0, (C), false, false)

// ---- async global->LDS (gfx1250 ASYNCcnt path), guarded for portability ----
#if defined(__has_builtin)
#if __has_builtin(__builtin_amdgcn_global_load_async_to_lds_b128) && \
    __has_builtin(__builtin_amdgcn_s_wait_asynccnt)
#define HAS_ASYNC_LDS 1
#endif
#endif
#ifndef HAS_ASYNC_LDS
#define HAS_ASYNC_LDS 0
#endif

#if HAS_ASYNC_LDS
typedef __attribute__((__vector_size__(16))) int async_v4i;
typedef __attribute__((address_space(1))) async_v4i* async_gp;
typedef __attribute__((address_space(3))) async_v4i* async_lp;
#endif

__device__ __forceinline__ void copy_g2l_b128(uint4* ldst, const uint4* gsrc,
                                              int n16, int tid, int nthreads) {
#if HAS_ASYNC_LDS
  for (int i = tid; i < n16; i += nthreads) {
    __builtin_amdgcn_global_load_async_to_lds_b128(
        (async_gp)(gsrc + i), (async_lp)(ldst + i), 0, 0);
  }
#else
  for (int i = tid; i < n16; i += nthreads) ldst[i] = gsrc[i];
#endif
}
__device__ __forceinline__ void wait_async_lds() {
#if HAS_ASYNC_LDS
  __builtin_amdgcn_s_wait_asynccnt(0);
#endif
}

__device__ __forceinline__ unsigned short f2bf(float f) {
  union { float f; unsigned int u; } a; a.f = f;
  unsigned int u = a.u;
  u += 0x7fffu + ((u >> 16) & 1u);           // round-to-nearest-even
  return (unsigned short)(u >> 16);
}
__device__ __forceinline__ unsigned int pack2bf(float lo, float hi) {
  return (unsigned int)f2bf(lo) | ((unsigned int)f2bf(hi) << 16);
}
__device__ __forceinline__ float sigm(float x) { return 1.0f / (1.0f + __expf(-x)); }

// ---- workspace layout --------------------------------------------------------
// floats at ws base:
#define GP0_OFF   16        // gn0 partials: [1024][2] floats
#define GS1_OFF   4096      // per-n partials: [256][4 waves][2]
#define GS2_OFF   8192
#define GS3_OFF   12288
// bytes:
#define W_OFF_B   65536u
// dword offsets inside swizzled-weight region:
#define W_SRU 0
#define W_C1  57344
#define W_C2  61440
#define W_C3  118784
#define W_CT  122880
#define W_TOTAL_DW 137216
#define U_OFF_B   (1u << 20)
#define Y_OFF_B   (U_OFF_B + (64u << 20))
#define Y2_OFF_B  (Y_OFF_B + (32u << 20))
// Y1 and Y3 alias the U region (U dead after scan, Y1 dead after conv2).

#define LSEQ 506
#define TSEQ 512

// ============================================================================
// gn0: global mean/var over all 8.4M elements (deterministic 2-stage reduce)
// ============================================================================
__global__ void gn0_partial(const float* __restrict__ x, float* __restrict__ S) {
  float s = 0.f, q = 0.f;
  size_t stride = (size_t)gridDim.x * blockDim.x;
  for (size_t i = (size_t)blockIdx.x * blockDim.x + threadIdx.x; i < (8388608u / 4); i += stride) {
    float4 v = ((const float4*)x)[i];
    s += (v.x + v.y) + (v.z + v.w);
    q += (v.x * v.x + v.y * v.y) + (v.z * v.z + v.w * v.w);
  }
  __shared__ float sh[16];
  #pragma unroll
  for (int o = 16; o; o >>= 1) { s += __shfl_xor(s, o, 32); q += __shfl_xor(q, o, 32); }
  int wv = threadIdx.x >> 5, lane = threadIdx.x & 31;
  if (lane == 0) { sh[wv] = s; sh[8 + wv] = q; }
  __syncthreads();
  if (threadIdx.x == 0) {
    float Sm = 0.f, Qm = 0.f;
    for (int i = 0; i < 8; ++i) { Sm += sh[i]; Qm += sh[8 + i]; }
    S[GP0_OFF + blockIdx.x * 2]     = Sm;
    S[GP0_OFF + blockIdx.x * 2 + 1] = Qm;
  }
}

__global__ void gn0_final(float* __restrict__ S) {
  __shared__ float sh[512];
  int t = threadIdx.x;  // 256 threads
  float s = 0.f, q = 0.f;
  for (int i = t; i < 1024; i += 256) { s += S[GP0_OFF + i * 2]; q += S[GP0_OFF + i * 2 + 1]; }
  sh[t] = s; sh[256 + t] = q;
  __syncthreads();
  for (int o = 128; o; o >>= 1) {
    if (t < o) { sh[t] += sh[t + o]; sh[256 + t] += sh[256 + t + o]; }
    __syncthreads();
  }
  if (t == 0) {
    const float cnt = 8388608.f;
    float mean = sh[0] / cnt;
    float var  = fmaxf(sh[256] / cnt - mean * mean, 0.f);
    S[2] = mean;
    S[3] = rsqrtf(var + 1e-6f);
  }
}

// ============================================================================
// prep_weights: swizzle all GEMM weights into per-lane bf16 WMMA B-fragments.
// B frags here (contiguous k, not ISA table order): lane = {hi=lane>>4, ncol=lane&15}; dword v packs
// K = kb*32 + 2v + 16*hi (even/odd pair). Reduction index j is permuted per op.
// ============================================================================
__device__ __forceinline__ float wsrc(int mat, int j, int n,
    const float* sru_w, const float* c1_w, const float* c2_w,
    const float* c3_w, const float* ct_w) {
  switch (mat) {
    case 0: { int k = j >> 6, c = j & 63, i = c * 7 + k, d = n >> 7, o = n & 127;
              return sru_w[(d * 448 + i) * 128 + o]; }                    // j = k*64+c
    case 1: return c1_w[n * 64 + j];                                      // j = cin
    case 2: { int k = j >> 7, c = j & 127;
              return c2_w[(n * 128 + c) * 7 + k]; }                       // j = k*128+c
    case 3: return c3_w[n * 128 + j];                                     // j = cin
    default:{ int jt = j >> 6, i = j & 63;
              return ct_w[(i * 64 + n) * 7 + jt]; }                       // j = tap*64+cin
  }
}

__global__ void prep_weights(const float* __restrict__ sru_w, const float* __restrict__ c1_w,
                             const float* __restrict__ c2_w, const float* __restrict__ c3_w,
                             const float* __restrict__ ct_w, unsigned int* __restrict__ W) {
  int idx = blockIdx.x * 256 + threadIdx.x;
  if (idx >= W_TOTAL_DW) return;
  int mat, rel, nnb;
  if      (idx < W_C1) { mat = 0; rel = idx;          nnb = 16; }
  else if (idx < W_C2) { mat = 1; rel = idx - W_C1;   nnb = 8;  }
  else if (idx < W_C3) { mat = 2; rel = idx - W_C2;   nnb = 8;  }
  else if (idx < W_CT) { mat = 3; rel = idx - W_C3;   nnb = 4;  }
  else                 { mat = 4; rel = idx - W_CT;   nnb = 4;  }
  int v = rel & 7, lane = (rel >> 3) & 31, frag = rel >> 8;
  int nb = frag % nnb, kb = frag / nnb;
  int hi = lane >> 4, ncol = lane & 15;
  int k0 = kb * 32 + 2 * v + 16 * hi;
  int n  = nb * 16 + ncol;
  float e0 = wsrc(mat, k0,     n, sru_w, c1_w, c2_w, c3_w, ct_w);
  float e1 = wsrc(mat, k0 + 1, n, sru_w, c1_w, c2_w, c3_w, ct_w);
  W[idx] = pack2bf(e0, e1);
}

// ---- A fragment K offset helper (contiguous k = 16*hi + 2v, same order as B) ----------------------
__device__ __forceinline__ int a_koff(int v, int hi) {
  return 2 * v + 16 * hi;
}

// ============================================================================
// sru_proj: per sequence n, U[d][n][l][o'] = A(506x448) * B(448x256), bf16 WMMA.
// A[l][j=k*64+c] = gn0(x)[c][l+k].  o' interleaves gate quarters for the scan.
// LDS: xT (512x64 bf16, 64KB) + all B fragments (229KB).
// ============================================================================
__global__ void __launch_bounds__(128) sru_proj(const float* __restrict__ x,
    const float* __restrict__ g0w, const float* __restrict__ g0b,
    const float* __restrict__ S, const unsigned int* __restrict__ W,
    float* __restrict__ U) {
  extern __shared__ unsigned char smem[];
  unsigned short* xT = (unsigned short*)smem;            // [512][64] bf16
  uint4* wl = (uint4*)(smem + 65536);                    // 14336 uint4
  const int n = blockIdx.x, tid = threadIdx.x;
  const int lane = tid & 31, wv = tid >> 5;
  const float mean = S[2], inv = S[3];

  // kick off async weight staging, overlap with activation conversion below
  copy_g2l_b128(wl, (const uint4*)(W + W_SRU), 14336, tid, 128);

  for (int task = tid; task < 64 * 128; task += 128) {
    int c = task >> 7, t4 = (task & 127) * 4;
    float4 vx = *(const float4*)(x + ((size_t)(c * 256 + n)) * TSEQ + t4);
    float sw = g0w[c] * inv, sb = g0b[c] - mean * sw;
    xT[(t4 + 0) * 64 + c] = f2bf(vx.x * sw + sb);
    xT[(t4 + 1) * 64 + c] = f2bf(vx.y * sw + sb);
    xT[(t4 + 2) * 64 + c] = f2bf(vx.z * sw + sb);
    xT[(t4 + 3) * 64 + c] = f2bf(vx.w * sw + sb);
  }
  wait_async_lds();
  __syncthreads();

  const int hi = lane >> 4, m = lane & 15, ncol = lane & 15;
  for (int mt = wv; mt < 32; mt += 4) {
    int l0 = mt * 16;
    f32x8 acc[16];
    const f32x8 zero = {0.f, 0.f, 0.f, 0.f, 0.f, 0.f, 0.f, 0.f};
    #pragma unroll
    for (int i = 0; i < 16; ++i) acc[i] = zero;

    #define SRU_LOADA(dst, KB) do {                                          \
      _Pragma("unroll")                                                      \
      for (int v_ = 0; v_ < 8; ++v_) {                                       \
        int j0_ = (KB) * 32 + a_koff(v_, hi);                                \
        int k_ = j0_ >> 6, c_ = j0_ & 63;                                    \
        int l_ = l0 + m;                                                     \
        (dst).u[v_] = (l_ < LSEQ)                                            \
          ? *(const unsigned int*)(xT + (l_ + k_) * 64 + c_) : 0u;           \
      } } while (0)
    #define SRU_LOADB(dst, NB) do {                                          \
      const uint4* p_ = wl + ((size_t)(kb * 16 + (NB)) * 32 + lane) * 2;     \
      (dst).q[0] = p_[0]; (dst).q[1] = p_[1]; } while (0)

    ABFrag a_cur; SRU_LOADA(a_cur, 0);
    for (int kb = 0; kb < 14; ++kb) {
      ABFrag a_nxt = a_cur;
      if (kb < 13) SRU_LOADA(a_nxt, kb + 1);
      ABFrag bb[4];
      SRU_LOADB(bb[0], 0); SRU_LOADB(bb[1], 1); SRU_LOADB(bb[2], 2);
      #pragma unroll
      for (int nb = 0; nb < 16; ++nb) {
        if (nb + 3 < 16) SRU_LOADB(bb[(nb + 3) & 3], nb + 3);
        acc[nb] = WMMA_BF16(acc[nb], a_cur.v, bb[nb & 3].v);
      }
      a_cur = a_nxt;
    }
    #undef SRU_LOADA
    #undef SRU_LOADB

    #pragma unroll
    for (int nb = 0; nb < 16; ++nb) {
      int og = nb * 16 + ncol;
      int d = og >> 7, o = og & 127;
      int oi = (o & 31) * 4 + (o >> 5);   // interleave gate quarters for scan
      float* Ub = U + ((size_t)(d * 256 + n)) * TSEQ * 128 + oi;
      #pragma unroll
      for (int r = 0; r < 8; ++r) {
        int l = l0 + r + 8 * hi;
        if (l < LSEQ) Ub[(size_t)l * 128] = acc[nb][r];
      }
    }
  }
}

// ============================================================================
// sru_scan: bidirectional SRU recurrence; one wave per (dir, n); lane = hidden.
// ============================================================================
__global__ void __launch_bounds__(256) sru_scan(const float* __restrict__ U,
    const float* __restrict__ sv, const float* __restrict__ sb,
    float* __restrict__ Y) {
  int g = blockIdx.x * 8 + (threadIdx.x >> 5);
  int lane = threadIdx.x & 31;
  int d = g >> 8, n = g & 255;
  const float* Ud = U + ((size_t)(d * 256 + n)) * TSEQ * 128;
  float vf = sv[(d * 2 + 0) * 32 + lane], vr = sv[(d * 2 + 1) * 32 + lane];
  float bf = sb[(d * 2 + 0) * 32 + lane], br = sb[(d * 2 + 1) * 32 + lane];
  float* Yd = Y + (size_t)n * TSEQ * 64 + d * 32 + lane;
  float c = 0.f;
  int p = d ? (LSEQ - 1) : 0;
  int sd = d ? -1 : 1;
  for (int s = 0; s < LSEQ; ++s) {
    const float* up = Ud + (size_t)p * 128 + lane * 4;
    __builtin_prefetch(up + sd * 8 * 128, 0, 1);       // global_prefetch_b8
    float4 u = *(const float4*)up;                      // (u0,u1,u2,u3)
    float f  = sigm(u.y + vf * c + bf);
    float cn = f * c + (1.f - f) * u.x;
    float r  = sigm(u.z + vr * c + br);
    float h  = r * cn + (1.f - r) * u.w;
    c = cn;
    Yd[(size_t)p * 64] = h;
    p += sd;
  }
}

// ============================================================================
// conv1 (1x1, 64->128): A from Y[n][l][64] in LDS; deterministic per-wave stats.
// ============================================================================
__global__ void __launch_bounds__(128) conv1(const float* __restrict__ Y,
    const unsigned int* __restrict__ W, const float* __restrict__ bias,
    float* __restrict__ Y1, float* __restrict__ GS) {
  extern __shared__ unsigned char smem[];
  unsigned short* yb = (unsigned short*)smem;          // [512][64] bf16
  uint4* wl = (uint4*)(smem + 65536);                  // 1024 uint4
  const int n = blockIdx.x, tid = threadIdx.x;
  const int lane = tid & 31, wv = tid >> 5;
  copy_g2l_b128(wl, (const uint4*)(W + W_C1), 1024, tid, 128);
  for (int task = tid; task < 512 * 32; task += 128) {
    int l = task >> 5, cp = (task & 31) * 2;
    unsigned int pk = 0u;
    if (l < LSEQ) {
      float2 v = *(const float2*)(Y + ((size_t)n * TSEQ + l) * 64 + cp);
      pk = pack2bf(v.x, v.y);
    }
    *(unsigned int*)(yb + l * 64 + cp) = pk;
  }
  wait_async_lds();
  __syncthreads();
  const int hi = lane >> 4, m = lane & 15, ncol = lane & 15;
  float s1 = 0.f, s2 = 0.f;
  for (int mt = wv; mt < 32; mt += 4) {
    int l0 = mt * 16;
    f32x8 acc[8];
    const f32x8 zero = {0.f, 0.f, 0.f, 0.f, 0.f, 0.f, 0.f, 0.f};
    #pragma unroll
    for (int i = 0; i < 8; ++i) acc[i] = zero;

    #define C1_LOADB(dst, KB, NB) do {                                       \
      const uint4* p_ = wl + ((size_t)((KB) * 8 + (NB)) * 32 + lane) * 2;    \
      (dst).q[0] = p_[0]; (dst).q[1] = p_[1]; } while (0)

    #pragma unroll
    for (int kb = 0; kb < 2; ++kb) {
      ABFrag a;
      #pragma unroll
      for (int v = 0; v < 8; ++v) {
        int c = kb * 32 + a_koff(v, hi);
        int l = l0 + m;
        a.u[v] = (l < LSEQ) ? *(const unsigned int*)(yb + l * 64 + c) : 0u;
      }
      ABFrag bb[4];
      C1_LOADB(bb[0], kb, 0); C1_LOADB(bb[1], kb, 1); C1_LOADB(bb[2], kb, 2);
      #pragma unroll
      for (int nb = 0; nb < 8; ++nb) {
        if (nb + 3 < 8) C1_LOADB(bb[(nb + 3) & 3], kb, nb + 3);
        acc[nb] = WMMA_BF16(acc[nb], a.v, bb[nb & 3].v);
      }
    }
    #undef C1_LOADB

    #pragma unroll
    for (int nb = 0; nb < 8; ++nb) {
      int o = nb * 16 + ncol;
      float bo = bias[o];
      #pragma unroll
      for (int r = 0; r < 8; ++r) {
        int l = l0 + r + 8 * hi;
        if (l < LSEQ) {
          float val = acc[nb][r] + bo;
          Y1[((size_t)n * TSEQ + l) * 128 + o] = val;
          s1 += val; s2 += val * val;
        }
      }
    }
  }
  #pragma unroll
  for (int o = 16; o; o >>= 1) { s1 += __shfl_xor(s1, o, 32); s2 += __shfl_xor(s2, o, 32); }
  if (lane == 0) { GS[(n * 4 + wv) * 2] = s1; GS[(n * 4 + wv) * 2 + 1] = s2; }
}

// ============================================================================
// conv2 (K=7 pad 3, 128->128): A[l][j=k*128+c] = y1[l+k-3][c]; B from L2 (hot).
// ============================================================================
__global__ void __launch_bounds__(128) conv2(const float* __restrict__ Y1,
    const unsigned int* __restrict__ W, const float* __restrict__ bias,
    float* __restrict__ Y2, float* __restrict__ GS) {
  extern __shared__ unsigned char smem[];
  unsigned short* yb = (unsigned short*)smem;          // [512][128] bf16
  const int n = blockIdx.x, tid = threadIdx.x;
  const int lane = tid & 31, wv = tid >> 5;
  for (int task = tid; task < 512 * 64; task += 128) {
    int l = task >> 6, cp = (task & 63) * 2;
    unsigned int pk = 0u;
    if (l < LSEQ) {
      float2 v = *(const float2*)(Y1 + ((size_t)n * TSEQ + l) * 128 + cp);
      pk = pack2bf(v.x, v.y);
    }
    *(unsigned int*)(yb + l * 128 + cp) = pk;
  }
  __syncthreads();
  const uint4* wg = (const uint4*)(W + W_C2);
  const int hi = lane >> 4, m = lane & 15, ncol = lane & 15;
  float s1 = 0.f, s2 = 0.f;
  for (int mt = wv; mt < 32; mt += 4) {
    int l0 = mt * 16;
    f32x8 acc[8];
    const f32x8 zero = {0.f, 0.f, 0.f, 0.f, 0.f, 0.f, 0.f, 0.f};
    #pragma unroll
    for (int i = 0; i < 8; ++i) acc[i] = zero;

    #define C2_LOADA(dst, KB) do {                                           \
      _Pragma("unroll")                                                      \
      for (int v_ = 0; v_ < 8; ++v_) {                                       \
        int j0_ = (KB) * 32 + a_koff(v_, hi);                                \
        int k_ = j0_ >> 7, c_ = j0_ & 127;                                   \
        int l_ = l0 + m;                                                     \
        int t_ = l_ + k_ - 3;                                                \
        (dst).u[v_] = (l_ < LSEQ && t_ >= 0 && t_ < LSEQ)                    \
          ? *(const unsigned int*)(yb + t_ * 128 + c_) : 0u;                 \
      } } while (0)
    #define C2_LOADB(dst, NB) do {                                           \
      const uint4* p_ = wg + ((size_t)(kb * 8 + (NB)) * 32 + lane) * 2;      \
      (dst).q[0] = p_[0]; (dst).q[1] = p_[1]; } while (0)

    ABFrag a_cur; C2_LOADA(a_cur, 0);
    for (int kb = 0; kb < 28; ++kb) {
      __builtin_prefetch(wg + ((size_t)((kb + 1) * 8) * 32 + lane) * 2, 0, 1);
      ABFrag a_nxt = a_cur;
      if (kb < 27) C2_LOADA(a_nxt, kb + 1);
      ABFrag bb[4];
      C2_LOADB(bb[0], 0); C2_LOADB(bb[1], 1); C2_LOADB(bb[2], 2);
      #pragma unroll
      for (int nb = 0; nb < 8; ++nb) {
        if (nb + 3 < 8) C2_LOADB(bb[(nb + 3) & 3], nb + 3);
        acc[nb] = WMMA_BF16(acc[nb], a_cur.v, bb[nb & 3].v);
      }
      a_cur = a_nxt;
    }
    #undef C2_LOADA
    #undef C2_LOADB

    #pragma unroll
    for (int nb = 0; nb < 8; ++nb) {
      int o = nb * 16 + ncol;
      float bo = bias[o];
      #pragma unroll
      for (int r = 0; r < 8; ++r) {
        int l = l0 + r + 8 * hi;
        if (l < LSEQ) {
          float val = acc[nb][r] + bo;
          Y2[((size_t)n * TSEQ + l) * 128 + o] = val;
          s1 += val; s2 += val * val;
        }
      }
    }
  }
  #pragma unroll
  for (int o = 16; o; o >>= 1) { s1 += __shfl_xor(s1, o, 32); s2 += __shfl_xor(s2, o, 32); }
  if (lane == 0) { GS[(n * 4 + wv) * 2] = s1; GS[(n * 4 + wv) * 2 + 1] = s2; }
}

// ============================================================================
// conv3 (1x1, 128->64)
// ============================================================================
__global__ void __launch_bounds__(128) conv3(const float* __restrict__ Y2,
    const unsigned int* __restrict__ W, const float* __restrict__ bias,
    float* __restrict__ Y3, float* __restrict__ GS) {
  extern __shared__ unsigned char smem[];
  unsigned short* yb = (unsigned short*)smem;          // [512][128] bf16
  uint4* wl = (uint4*)(smem + 131072);                 // 1024 uint4
  const int n = blockIdx.x, tid = threadIdx.x;
  const int lane = tid & 31, wv = tid >> 5;
  copy_g2l_b128(wl, (const uint4*)(W + W_C3), 1024, tid, 128);
  for (int task = tid; task < 512 * 64; task += 128) {
    int l = task >> 6, cp = (task & 63) * 2;
    unsigned int pk = 0u;
    if (l < LSEQ) {
      float2 v = *(const float2*)(Y2 + ((size_t)n * TSEQ + l) * 128 + cp);
      pk = pack2bf(v.x, v.y);
    }
    *(unsigned int*)(yb + l * 128 + cp) = pk;
  }
  wait_async_lds();
  __syncthreads();
  const int hi = lane >> 4, m = lane & 15, ncol = lane & 15;
  float s1 = 0.f, s2 = 0.f;
  for (int mt = wv; mt < 32; mt += 4) {
    int l0 = mt * 16;
    f32x8 acc[4];
    const f32x8 zero = {0.f, 0.f, 0.f, 0.f, 0.f, 0.f, 0.f, 0.f};
    #pragma unroll
    for (int i = 0; i < 4; ++i) acc[i] = zero;

    #define C3_LOADB(dst, KB, NB) do {                                       \
      const uint4* p_ = wl + ((size_t)((KB) * 4 + (NB)) * 32 + lane) * 2;    \
      (dst).q[0] = p_[0]; (dst).q[1] = p_[1]; } while (0)

    #pragma unroll
    for (int kb = 0; kb < 4; ++kb) {
      ABFrag a;
      #pragma unroll
      for (int v = 0; v < 8; ++v) {
        int c = kb * 32 + a_koff(v, hi);
        int l = l0 + m;
        a.u[v] = (l < LSEQ) ? *(const unsigned int*)(yb + l * 128 + c) : 0u;
      }
      ABFrag bb[4];
      C3_LOADB(bb[0], kb, 0); C3_LOADB(bb[1], kb, 1); C3_LOADB(bb[2], kb, 2);
      #pragma unroll
      for (int nb = 0; nb < 4; ++nb) {
        if (nb + 3 < 4) C3_LOADB(bb[(nb + 3) & 3], kb, nb + 3);
        acc[nb] = WMMA_BF16(acc[nb], a.v, bb[nb & 3].v);
      }
    }
    #undef C3_LOADB

    #pragma unroll
    for (int nb = 0; nb < 4; ++nb) {
      int o = nb * 16 + ncol;
      float bo = bias[o];
      #pragma unroll
      for (int r = 0; r < 8; ++r) {
        int l = l0 + r + 8 * hi;
        if (l < LSEQ) {
          float val = acc[nb][r] + bo;
          Y3[((size_t)n * TSEQ + l) * 64 + o] = val;
          s1 += val; s2 += val * val;
        }
      }
    }
  }
  #pragma unroll
  for (int o = 16; o; o >>= 1) { s1 += __shfl_xor(s1, o, 32); s2 += __shfl_xor(s2, o, 32); }
  if (lane == 0) { GS[(n * 4 + wv) * 2] = s1; GS[(n * 4 + wv) * 2 + 1] = s2; }
}

// ============================================================================
// norm_apply: per-n group norm + prelu (+ optional residual add)
// ============================================================================
__global__ void norm_apply(float* __restrict__ buf, const float* __restrict__ GS,
                           const float* __restrict__ gw, const float* __restrict__ gb,
                           const float* __restrict__ addb, int C, float invcnt) {
  size_t idx = (size_t)blockIdx.x * 256 + threadIdx.x;
  size_t total = (size_t)256 * LSEQ * C;
  if (idx >= total) return;
  int o = (int)(idx % C);
  size_t r = idx / C;
  int l = (int)(r % LSEQ);
  int n = (int)(r / LSEQ);
  float s = GS[n * 8 + 0] + GS[n * 8 + 2] + GS[n * 8 + 4] + GS[n * 8 + 6];
  float q = GS[n * 8 + 1] + GS[n * 8 + 3] + GS[n * 8 + 5] + GS[n * 8 + 7];
  float mean = s * invcnt;
  float var  = fmaxf(q * invcnt - mean * mean, 0.f);
  float inv  = rsqrtf(var + 1e-6f);
  size_t off = ((size_t)n * TSEQ + l) * C + o;
  float v = (buf[off] - mean) * inv * gw[o] + gb[o];
  v = (v >= 0.f) ? v : 0.25f * v;
  if (addb) v += addb[off];
  buf[off] = v;
}

// ============================================================================
// convT (transposed conv 64->64, K=7): out[t] = sum_j y3[t-j]*w[i][o][j].
// Stages the (64 x 512) tile in LDS (stride 513), then coalesced store + skip.
// ============================================================================
__global__ void __launch_bounds__(128) convT(const float* __restrict__ Y3,
    const unsigned int* __restrict__ W, const float* __restrict__ ctb,
    const float* __restrict__ xin, float* __restrict__ outp) {
  extern __shared__ unsigned char smem[];
  unsigned short* yb = (unsigned short*)smem;            // [512][64] bf16
  uint4* wl = (uint4*)(smem + 65536);                    // 3584 uint4
  float* stage = (float*)(smem + 65536 + 57344);         // [64][513] f32
  const int n = blockIdx.x, tid = threadIdx.x;
  const int lane = tid & 31, wv = tid >> 5;
  copy_g2l_b128(wl, (const uint4*)(W + W_CT), 3584, tid, 128);
  for (int task = tid; task < 512 * 32; task += 128) {
    int l = task >> 5, cp = (task & 31) * 2;
    unsigned int pk = 0u;
    if (l < LSEQ) {
      float2 v = *(const float2*)(Y3 + ((size_t)n * TSEQ + l) * 64 + cp);
      pk = pack2bf(v.x, v.y);
    }
    *(unsigned int*)(yb + l * 64 + cp) = pk;             // rows >= 506 zeroed
  }
  wait_async_lds();
  __syncthreads();
  const int hi = lane >> 4, m = lane & 15, ncol = lane & 15;
  for (int mt = wv; mt < 32; mt += 4) {
    int t0 = mt * 16;
    f32x8 acc[4];
    const f32x8 zero = {0.f, 0.f, 0.f, 0.f, 0.f, 0.f, 0.f, 0.f};
    #pragma unroll
    for (int i = 0; i < 4; ++i) acc[i] = zero;

    #define CT_LOADA(dst, KB) do {                                           \
      _Pragma("unroll")                                                      \
      for (int v_ = 0; v_ < 8; ++v_) {                                       \
        int j0_ = (KB) * 32 + a_koff(v_, hi);                                \
        int jt_ = j0_ >> 6, i_ = j0_ & 63;                                   \
        int t_ = t0 + m;                                                     \
        int l_ = t_ - jt_;                                                   \
        (dst).u[v_] = (l_ >= 0)                                              \
          ? *(const unsigned int*)(yb + l_ * 64 + i_) : 0u;                  \
      } } while (0)
    #define CT_LOADB(dst, NB) do {                                           \
      const uint4* p_ = wl + ((size_t)(kb * 4 + (NB)) * 32 + lane) * 2;      \
      (dst).q[0] = p_[0]; (dst).q[1] = p_[1]; } while (0)

    ABFrag a_cur; CT_LOADA(a_cur, 0);
    for (int kb = 0; kb < 14; ++kb) {
      ABFrag a_nxt = a_cur;
      if (kb < 13) CT_LOADA(a_nxt, kb + 1);
      ABFrag bb[4];
      CT_LOADB(bb[0], 0); CT_LOADB(bb[1], 1); CT_LOADB(bb[2], 2);
      #pragma unroll
      for (int nb = 0; nb < 4; ++nb) {
        if (nb + 3 < 4) CT_LOADB(bb[(nb + 3) & 3], nb + 3);
        acc[nb] = WMMA_BF16(acc[nb], a_cur.v, bb[nb & 3].v);
      }
      a_cur = a_nxt;
    }
    #undef CT_LOADA
    #undef CT_LOADB

    #pragma unroll
    for (int nb = 0; nb < 4; ++nb) {
      int o = nb * 16 + ncol;
      #pragma unroll
      for (int r = 0; r < 8; ++r) {
        int t = t0 + r + 8 * hi;
        stage[o * 513 + t] = acc[nb][r];
      }
    }
  }
  __syncthreads();
  for (int task = tid; task < 64 * 128; task += 128) {
    int o = task >> 7, t4 = (task & 127) * 4;
    float bo = ctb[o];
    size_t goff = ((size_t)(o * 256 + n)) * TSEQ + t4;
    float4 xi = *(const float4*)(xin + goff);
    float4 rv;
    rv.x = stage[o * 513 + t4 + 0] + bo + xi.x;
    rv.y = stage[o * 513 + t4 + 1] + bo + xi.y;
    rv.z = stage[o * 513 + t4 + 2] + bo + xi.z;
    rv.w = stage[o * 513 + t4 + 3] + bo + xi.w;
    *(float4*)(outp + goff) = rv;
  }
}

// ============================================================================
extern "C" void kernel_launch(void* const* d_in, const int* in_sizes, int n_in,
                              void* d_out, int out_size, void* d_ws, size_t ws_size,
                              hipStream_t stream) {
  (void)in_sizes; (void)n_in; (void)out_size; (void)ws_size;
  const float* x    = (const float*)d_in[0];
  const float* g0w  = (const float*)d_in[1];
  const float* g0b  = (const float*)d_in[2];
  const float* sruw = (const float*)d_in[3];
  const float* sruv = (const float*)d_in[4];
  const float* srub = (const float*)d_in[5];
  const float* c1w  = (const float*)d_in[6];
  const float* c1b  = (const float*)d_in[7];
  const float* g1w  = (const float*)d_in[8];
  const float* g1b  = (const float*)d_in[9];
  const float* c2w  = (const float*)d_in[10];
  const float* c2b  = (const float*)d_in[11];
  const float* g2w  = (const float*)d_in[12];
  const float* g2b  = (const float*)d_in[13];
  const float* c3w  = (const float*)d_in[14];
  const float* c3b  = (const float*)d_in[15];
  const float* g3w  = (const float*)d_in[16];
  const float* g3b  = (const float*)d_in[17];
  const float* ctw  = (const float*)d_in[18];
  const float* ctb  = (const float*)d_in[19];
  float* out = (float*)d_out;
  unsigned char* ws = (unsigned char*)d_ws;
  float* S = (float*)ws;
  unsigned int* W = (unsigned int*)(ws + W_OFF_B);
  float* U  = (float*)(ws + U_OFF_B);
  float* Y  = (float*)(ws + Y_OFF_B);
  float* Y1 = (float*)(ws + U_OFF_B);    // alias U (dead after scan)
  float* Y2 = (float*)(ws + Y2_OFF_B);
  float* Y3 = (float*)(ws + U_OFF_B);    // alias Y1 (dead after conv2)

  gn0_partial<<<1024, 256, 0, stream>>>(x, S);
  gn0_final<<<1, 256, 0, stream>>>(S);
  prep_weights<<<(W_TOTAL_DW + 255) / 256, 256, 0, stream>>>(sruw, c1w, c2w, c3w, ctw, W);
  sru_proj<<<256, 128, 294912, stream>>>(x, g0w, g0b, S, W, U);
  sru_scan<<<64, 256, 0, stream>>>(U, sruv, srub, Y);
  conv1<<<256, 128, 81920, stream>>>(Y, W, c1b, Y1, S + GS1_OFF);
  norm_apply<<<(256 * LSEQ * 128 + 255) / 256, 256, 0, stream>>>(
      Y1, S + GS1_OFF, g1w, g1b, nullptr, 128, 1.f / (128.f * (float)LSEQ));
  conv2<<<256, 128, 131072, stream>>>(Y1, W, c2b, Y2, S + GS2_OFF);
  norm_apply<<<(256 * LSEQ * 128 + 255) / 256, 256, 0, stream>>>(
      Y2, S + GS2_OFF, g2w, g2b, nullptr, 128, 1.f / (128.f * (float)LSEQ));
  conv3<<<256, 128, 147456, stream>>>(Y2, W, c3b, Y3, S + GS3_OFF);
  norm_apply<<<(256 * LSEQ * 64 + 255) / 256, 256, 0, stream>>>(
      Y3, S + GS3_OFF, g3w, g3b, Y, 64, 1.f / (64.f * (float)LSEQ));
  convT<<<256, 128, 254208, stream>>>(Y3, W, ctb, x, out);
}
